// PGAU_48627619726041
// MI455X (gfx1250) — matrix operand order repaired and checked by reading, not compiled, not run on hardware
//
#include <hip/hip_runtime.h>
#include <hip/hip_bf16.h>
#include <math.h>

// ---------------------------------------------------------------------------
// Problem constants (from reference)
// ---------------------------------------------------------------------------
#define BATCH 32
#define LSEQ  1024
#define DIN   512
#define DOUT  512
#define EDIM  1024
#define SDIM  128
#define UVN   (2 * EDIM + SDIM)   // 2176
#define MTOT  (BATCH * LSEQ)      // 32768
#define INV_SQRT_S 0.08838834764831845f  // 1/sqrt(128)

typedef __attribute__((ext_vector_type(16))) _Float16 v16h;
typedef __attribute__((ext_vector_type(8)))  _Float16 v8h;
typedef __attribute__((ext_vector_type(8)))  float    v8f;

// Async global->LDS (CDNA5): guarded so the file compiles on any toolchain.
#if defined(__has_builtin)
#if __has_builtin(__builtin_amdgcn_global_load_async_to_lds_b128) && \
    __has_builtin(__builtin_amdgcn_s_wait_asynccnt)
#define HAVE_ASYNC_LDS 1
#endif
#endif
#ifndef HAVE_ASYNC_LDS
#define HAVE_ASYNC_LDS 0
#endif

union FragH { v16h f; v8h h8[2]; };

__device__ inline v8f vzero() {
    v8f z;
#pragma unroll
    for (int i = 0; i < 8; ++i) z[i] = 0.f;
    return z;
}

__device__ inline v8f wmma32(v16h a, v16h b, v8f c) {
    return __builtin_amdgcn_wmma_f32_16x16x32_f16(
        false, a, false, b, (short)0, c, false, false);
}

// A-operand (16x32 [M][K] tile): lanes 0-15 M=lane, K0..7 / K16..23; lanes 16-31 +8 on K.
__device__ inline v16h frag_a(const _Float16* base, int row, int stride, int koff, int hi) {
    FragH r;
    const _Float16* p = base + row * stride + koff + (hi ? 8 : 0);
    r.h8[0] = *(const v8h*)(p);
    r.h8[1] = *(const v8h*)(p + 16);
    return r.f;
}

// B-operand (32x16 stored as [N][K] tile): lanes 0-15 N=lane, K0..7 / K16..23; lanes 16-31 +8 on K.
__device__ inline v16h frag_b(const _Float16* base, int nrow, int stride, int koff, int hi) {
    FragH r;
    const _Float16* p = base + nrow * stride + koff + (hi ? 8 : 0);
    r.h8[0] = *(const v8h*)(p);
    r.h8[1] = *(const v8h*)(p + 16);
    return r.f;
}

__device__ inline float fast_silu(float x) {
    return x * __builtin_amdgcn_rcpf(1.f + __expf(-x));
}

#if HAVE_ASYNC_LDS
// Signature per compiler diagnostic: (v4i AS1*, v4i AS3*, imm offset, imm cpol)
typedef __attribute__((__vector_size__(4 * sizeof(int)))) int async_v4i;
typedef async_v4i __attribute__((address_space(1)))* async_gptr;
typedef async_v4i __attribute__((address_space(3)))* async_lptr;

__device__ inline void async_ld128(_Float16* ldsDst, const _Float16* gSrc) {
    __builtin_amdgcn_global_load_async_to_lds_b128(
        (async_gptr)(gSrc), (async_lptr)(ldsDst), 0, 0);
}
#endif

// ---------------------------------------------------------------------------
// Kernel 1: RMS-style norm + g with token-shift scatter.
// ---------------------------------------------------------------------------
__global__ void pgau_norm_shift(const float* __restrict__ x,
                                const float* __restrict__ g,
                                _Float16* __restrict__ xn) {
    const int row = blockIdx.x;            // b*LSEQ + l
    const int l   = row & (LSEQ - 1);
    const int tid = threadIdx.x;           // 256 threads, 2 elems each
    const float2 val = *(const float2*)(x + (size_t)row * DIN + tid * 2);
    float ss = val.x * val.x + val.y * val.y;
#pragma unroll
    for (int off = 16; off > 0; off >>= 1) ss += __shfl_xor(ss, off, 32);
    __shared__ float red[8];
    if ((tid & 31) == 0) red[tid >> 5] = ss;
    __syncthreads();
    float tot = 0.f;
#pragma unroll
    for (int i = 0; i < 8; ++i) tot += red[i];
    const float nrm   = sqrtf(tot * (1.0f / DIN));
    const float scale = g[0] / fmaxf(nrm, 1e-5f);
    const _Float16 h0 = (_Float16)(val.x * scale);
    const _Float16 h1 = (_Float16)(val.y * scale);
    const int c0 = tid * 2;
    if (c0 < DIN / 2) {
        if (l + 1 < LSEQ) {
            xn[(size_t)(row + 1) * DIN + c0]     = h0;
            xn[(size_t)(row + 1) * DIN + c0 + 1] = h1;
        }
        if (l == 0) {
            xn[(size_t)row * DIN + c0]     = (_Float16)0.f;
            xn[(size_t)row * DIN + c0 + 1] = (_Float16)0.f;
        }
    } else {
        xn[(size_t)row * DIN + c0]     = h0;
        xn[(size_t)row * DIN + c0 + 1] = h1;
    }
}

// ---------------------------------------------------------------------------
// Kernel 2: f32 -> f16 conversion (weights)
// ---------------------------------------------------------------------------
__global__ void pgau_f32_to_f16(const float* __restrict__ in,
                                _Float16* __restrict__ out, int n) {
    int i = blockIdx.x * 256 + threadIdx.x;
    if (i < n) out[i] = (_Float16)in[i];
}

// ---------------------------------------------------------------------------
// Triple-buffered WMMA main loop: 128x128 tile, 32-wide K slabs, two slabs
// in flight via ASYNCcnt (in-order completion => wait <= 4 releases the
// older slab while the newest keeps flying). Fully unrolled so the buffer
// rotation folds into ds immediate offsets.
// ---------------------------------------------------------------------------
#if HAVE_ASYNC_LDS
#define ASYNC_STAGE(kb, Ad, Bd)                      \
    do {                                             \
        async_ld128((Ad) + sOff,     gA + (kb));     \
        async_ld128((Ad) + sOff + 8, gA + (kb) + 8); \
        async_ld128((Bd) + sOff,     gB + (kb));     \
        async_ld128((Bd) + sOff + 8, gB + (kb) + 8); \
    } while (0)

#define GEMM_MAINLOOP(KTOT, LDA)                                                   \
    const int sRow = tid >> 1;                                                     \
    const int sSeg = (tid & 1) * 16;                                               \
    const int sOff = sRow * 40 + sSeg;                                             \
    const _Float16* gA = Ah + (size_t)(m0 + sRow) * (LDA) + sSeg;                  \
    const _Float16* gB = Bh + (size_t)(n0 + sRow) * (LDA) + sSeg;                  \
    auto compute = [&](const _Float16* Ac, const _Float16* Bc) {                   \
        v16h af[2], bf[4];                                                         \
        _Pragma("unroll")                                                          \
        for (int mi = 0; mi < 2; ++mi)                                             \
            af[mi] = frag_a(Ac, wm * 32 + mi * 16 + ln, 40, 0, hi);                \
        _Pragma("unroll")                                                          \
        for (int ni = 0; ni < 4; ++ni)                                             \
            bf[ni] = frag_b(Bc, wn * 64 + ni * 16 + ln, 40, 0, hi);                \
        _Pragma("unroll")                                                          \
        for (int mi = 0; mi < 2; ++mi)                                             \
            _Pragma("unroll")                                                      \
            for (int ni = 0; ni < 4; ++ni)                                         \
                acc[mi][ni] = wmma32(af[mi], bf[ni], acc[mi][ni]);                 \
    };                                                                             \
    ASYNC_STAGE(0, As[0], Bs[0]);                                                  \
    ASYNC_STAGE(32, As[1], Bs[1]);                                                 \
    __builtin_amdgcn_s_wait_asynccnt(4);  /* slab 0 landed, slab 1 in flight */    \
    __syncthreads();                                                               \
    _Pragma("unroll")                                                              \
    for (int s = 0; s < (KTOT) / 32; ++s) {                                        \
        const int  kb     = s * 32;                                                \
        const bool staged = (kb + 64) < (KTOT);                                    \
        const int  cur    = s % 3;                                                 \
        const int  nx2    = (s + 2) % 3;                                           \
        if (staged) ASYNC_STAGE(kb + 64, As[nx2], Bs[nx2]);                        \
        compute(As[cur], Bs[cur]);                                                 \
        if (staged) __builtin_amdgcn_s_wait_asynccnt(4);                           \
        else        __builtin_amdgcn_s_wait_asynccnt(0);                           \
        __syncthreads();                                                           \
    }
#else
#define GEMM_MAINLOOP(KTOT, LDA)                                                   \
    const int sRow = tid >> 1;                                                     \
    const int sSeg = (tid & 1) * 16;                                               \
    const int sOff = sRow * 40 + sSeg;                                             \
    const _Float16* gA = Ah + (size_t)(m0 + sRow) * (LDA) + sSeg;                  \
    const _Float16* gB = Bh + (size_t)(n0 + sRow) * (LDA) + sSeg;                  \
    v8h ra0, ra1, rb0, rb1;                                                        \
    auto reg_load = [&](int kb) {                                                  \
        ra0 = *(const v8h*)(gA + kb);  ra1 = *(const v8h*)(gA + kb + 8);           \
        rb0 = *(const v8h*)(gB + kb);  rb1 = *(const v8h*)(gB + kb + 8);           \
    };                                                                             \
    auto reg_store = [&](_Float16* Ad, _Float16* Bd) {                             \
        *(v8h*)(Ad + sOff) = ra0;  *(v8h*)(Ad + sOff + 8) = ra1;                   \
        *(v8h*)(Bd + sOff) = rb0;  *(v8h*)(Bd + sOff + 8) = rb1;                   \
    };                                                                             \
    auto compute = [&](const _Float16* Ac, const _Float16* Bc) {                   \
        v16h af[2], bf[4];                                                         \
        _Pragma("unroll")                                                          \
        for (int mi = 0; mi < 2; ++mi)                                             \
            af[mi] = frag_a(Ac, wm * 32 + mi * 16 + ln, 40, 0, hi);                \
        _Pragma("unroll")                                                          \
        for (int ni = 0; ni < 4; ++ni)                                             \
            bf[ni] = frag_b(Bc, wn * 64 + ni * 16 + ln, 40, 0, hi);                \
        _Pragma("unroll")                                                          \
        for (int mi = 0; mi < 2; ++mi)                                             \
            _Pragma("unroll")                                                      \
            for (int ni = 0; ni < 4; ++ni)                                         \
                acc[mi][ni] = wmma32(af[mi], bf[ni], acc[mi][ni]);                 \
    };                                                                             \
    reg_load(0);                                                                   \
    reg_store(As[0], Bs[0]);                                                       \
    __syncthreads();                                                               \
    _Pragma("unroll")                                                              \
    for (int s = 0; s < (KTOT) / 32; ++s) {                                        \
        const int  kb = s * 32;                                                    \
        const bool nx = (kb + 32) < (KTOT);                                        \
        if (nx) reg_load(kb + 32);                                                 \
        compute(As[s % 3], Bs[s % 3]);                                             \
        if (nx) reg_store(As[(s + 1) % 3], Bs[(s + 1) % 3]);                       \
        __syncthreads();                                                           \
    }
#endif

// ---------------------------------------------------------------------------
// Kernel 3: uv = silu(xn @ uv_w^T); cols -> u (f16), v (f16), base (f32)
// ---------------------------------------------------------------------------
__global__ __launch_bounds__(256) void pgau_gemm_uv(
    const _Float16* __restrict__ Ah,   // [MTOT, 512]
    const _Float16* __restrict__ Bh,   // [2176, 512]
    _Float16* __restrict__ u_h, _Float16* __restrict__ v_h,
    float* __restrict__ base_f) {
    __shared__ _Float16 As[3][128 * 40];
    __shared__ _Float16 Bs[3][128 * 40];
    const int tid = threadIdx.x, wave = tid >> 5, lane = tid & 31;
    const int wm = wave & 3, wn = wave >> 2;
    const int hi = lane >> 4, ln = lane & 15;
    const int m0 = blockIdx.x * 128, n0 = blockIdx.y * 128;

    v8f acc[2][4];
#pragma unroll
    for (int i = 0; i < 2; ++i)
#pragma unroll
        for (int j = 0; j < 4; ++j) acc[i][j] = vzero();

    GEMM_MAINLOOP(512, 512)

    // Block-uniform routing: n0 is a multiple of 128, region bounds are too.
    const int mbase = m0 + wm * 32;
    const int nbase = n0 + wn * 64;
    if (n0 < EDIM) {
#pragma unroll
        for (int mi = 0; mi < 2; ++mi)
#pragma unroll
            for (int ni = 0; ni < 4; ++ni)
#pragma unroll
                for (int e = 0; e < 8; ++e) {
                    const int gm = mbase + mi * 16 + hi * 8 + e;
                    const int gn = nbase + ni * 16 + ln;
                    u_h[(size_t)gm * EDIM + gn] = (_Float16)fast_silu(acc[mi][ni][e]);
                }
    } else if (n0 < 2 * EDIM) {
#pragma unroll
        for (int mi = 0; mi < 2; ++mi)
#pragma unroll
            for (int ni = 0; ni < 4; ++ni)
#pragma unroll
                for (int e = 0; e < 8; ++e) {
                    const int gm = mbase + mi * 16 + hi * 8 + e;
                    const int gn = nbase + ni * 16 + ln - EDIM;
                    v_h[(size_t)gm * EDIM + gn] = (_Float16)fast_silu(acc[mi][ni][e]);
                }
    } else {
#pragma unroll
        for (int mi = 0; mi < 2; ++mi)
#pragma unroll
            for (int ni = 0; ni < 4; ++ni)
#pragma unroll
                for (int e = 0; e < 8; ++e) {
                    const int gm = mbase + mi * 16 + hi * 8 + e;
                    const int gn = nbase + ni * 16 + ln - 2 * EDIM;
                    base_f[(size_t)gm * SDIM + gn] = fast_silu(acc[mi][ni][e]);
                }
    }
}

// ---------------------------------------------------------------------------
// Kernel 4: base -> q, k : gamma/beta then RoPE over sequence (half = 64)
// ---------------------------------------------------------------------------
__global__ void pgau_rope_qk(const float* __restrict__ base_f,
                             const float* __restrict__ gamma,
                             const float* __restrict__ beta,
                             _Float16* __restrict__ q, _Float16* __restrict__ k) {
    const int row = blockIdx.x;
    const int l   = row & (LSEQ - 1);
    const int j   = threadIdx.x;          // 0..63
    const float b1 = base_f[(size_t)row * SDIM + j];
    const float b2 = base_f[(size_t)row * SDIM + 64 + j];
    const float fr = __powf(10000.f, (float)j * (1.f / 64.f));
    float sn, cs;
    __sincosf((float)l * fr, &sn, &cs);
    {
        const float y1 = b1 * gamma[j]      + beta[j];
        const float y2 = b2 * gamma[64 + j] + beta[64 + j];
        q[(size_t)row * SDIM + j]      = (_Float16)(y1 * cs - y2 * sn);
        q[(size_t)row * SDIM + 64 + j] = (_Float16)(y2 * cs + y1 * sn);
    }
    {
        const float y1 = b1 * gamma[128 + j] + beta[128 + j];
        const float y2 = b2 * gamma[192 + j] + beta[192 + j];
        k[(size_t)row * SDIM + j]      = (_Float16)(y1 * cs - y2 * sn);
        k[(size_t)row * SDIM + 64 + j] = (_Float16)(y2 * cs + y1 * sn);
    }
}

// ---------------------------------------------------------------------------
// Kernel 5: fused attention with register prefetch of next K/V chunk.
// Block = (64 L-rows) x (256 E-cols) x batch.
// ---------------------------------------------------------------------------
__global__ __launch_bounds__(256) void pgau_attn(
    const _Float16* __restrict__ q, const _Float16* __restrict__ k,
    const _Float16* __restrict__ v, const _Float16* __restrict__ u,
    const float* __restrict__ wbias, _Float16* __restrict__ attn) {
    const int b  = blockIdx.z;
    const int l0 = blockIdx.x * 64;
    const int e0 = blockIdx.y * 256;
    const int tid = threadIdx.x, wave = tid >> 5, lane = tid & 31;
    const int hi = lane >> 4, ln = lane & 15;
    const int wA = wave & 3, cA = wave >> 2;
    const int wB = wave & 3, cB = wave >> 2;

    __shared__ _Float16 Qs[64 * 136];
    __shared__ _Float16 Ks[64 * 136];
    __shared__ _Float16 Vt[256 * 72];          // transposed: [e][m]
    __shared__ _Float16 Ss[64 * 72];           // relu^2 scores, f16

    const _Float16* qb = q + (size_t)b * LSEQ * SDIM;
    const _Float16* kb = k + (size_t)b * LSEQ * SDIM;
    const _Float16* vb = v + (size_t)b * LSEQ * EDIM;

    for (int idx = tid; idx < 64 * 16; idx += 256) {
        const int r = idx >> 4, c8 = (idx & 15) * 8;
        *(v8h*)(&Qs[r * 136 + c8]) = *(const v8h*)(qb + (size_t)(l0 + r) * SDIM + c8);
    }

    // register staging of the current K/V chunk
    v8h kreg[4], vreg[8];
    auto chunk_load = [&](int mb) {
#pragma unroll
        for (int i = 0; i < 4; ++i) {
            const int idx = tid + i * 256;
            const int r = idx >> 4, c8 = (idx & 15) * 8;
            kreg[i] = *(const v8h*)(kb + (size_t)(mb + r) * SDIM + c8);
        }
#pragma unroll
        for (int i = 0; i < 8; ++i) {
            const int idx = tid + i * 256;
            const int m = idx >> 5, e8 = (idx & 31) * 8;
            vreg[i] = *(const v8h*)(vb + (size_t)(mb + m) * EDIM + e0 + e8);
        }
    };
    auto chunk_store = [&]() {
#pragma unroll
        for (int i = 0; i < 4; ++i) {
            const int idx = tid + i * 256;
            const int r = idx >> 4, c8 = (idx & 15) * 8;
            *(v8h*)(&Ks[r * 136 + c8]) = kreg[i];
        }
#pragma unroll
        for (int i = 0; i < 8; ++i) {
            const int idx = tid + i * 256;
            const int m = idx >> 5, e8 = (idx & 31) * 8;
#pragma unroll
            for (int j = 0; j < 8; ++j) Vt[(e8 + j) * 72 + m] = vreg[i][j];
        }
    };

    v8f acc[8];
#pragma unroll
    for (int i = 0; i < 8; ++i) acc[i] = vzero();

    chunk_load(0);
    for (int mb = 0; mb < LSEQ; mb += 64) {
        chunk_store();                        // prior iteration's tail barrier protects LDS
        if (mb + 64 < LSEQ) chunk_load(mb + 64);  // prefetch next chunk
        __syncthreads();

        // phase A: S = relu^2((Q K^T + w)/sqrt(S))
        v8f sa[2];
        sa[0] = vzero(); sa[1] = vzero();
#pragma unroll
        for (int ks = 0; ks < 4; ++ks) {
            v16h af = frag_a(Qs, wA * 16 + ln, 136, ks * 32, hi);
#pragma unroll
            for (int ci = 0; ci < 2; ++ci) {
                v16h bf = frag_b(Ks, (cA * 2 + ci) * 16 + ln, 136, ks * 32, hi);
                sa[ci] = wmma32(af, bf, sa[ci]);
            }
        }
#pragma unroll
        for (int ci = 0; ci < 2; ++ci) {
            const int rB = wA * 16 + hi * 8;
            const int nB = (cA * 2 + ci) * 16 + ln;
            const int gm = mb + nB;
#pragma unroll
            for (int e = 0; e < 8; ++e) {
                const int gl = l0 + rB + e;
                float t = (sa[ci][e] + wbias[gm - gl + (LSEQ - 1)]) * INV_SQRT_S;
                t = fmaxf(t, 0.f);
                Ss[(rB + e) * 72 + nB] = (_Float16)(t * t);
            }
        }
        __syncthreads();

        // phase B: acc += Ss @ Vt
#pragma unroll
        for (int ks = 0; ks < 2; ++ks) {
            v16h af = frag_a(Ss, wB * 16 + ln, 72, ks * 32, hi);
#pragma unroll
            for (int ni = 0; ni < 8; ++ni) {
                v16h bf = frag_b(Vt, cB * 128 + ni * 16 + ln, 72, ks * 32, hi);
                acc[ni] = wmma32(af, bf, acc[ni]);
            }
        }
        __syncthreads();
    }

    // epilogue: gate by u, store f16
#pragma unroll
    for (int ni = 0; ni < 8; ++ni)
#pragma unroll
        for (int e = 0; e < 8; ++e) {
            const int gl = l0 + wB * 16 + hi * 8 + e;
            const int ge = e0 + cB * 128 + ni * 16 + ln;
            const size_t off = ((size_t)b * LSEQ + gl) * EDIM + ge;
            attn[off] = (_Float16)(acc[ni][e] * (float)u[off]);
        }
}

// ---------------------------------------------------------------------------
// Kernel 6: out = x*res_scale + (attn @ o_w^T)*layer_scale
// ---------------------------------------------------------------------------
__global__ __launch_bounds__(256) void pgau_gemm_out(
    const _Float16* __restrict__ Ah,   // [MTOT, 1024]
    const _Float16* __restrict__ Bh,   // [512, 1024]
    const float* __restrict__ xin, const float* __restrict__ res_scale,
    const float* __restrict__ layer_scale, float* __restrict__ out) {
    __shared__ _Float16 As[3][128 * 40];
    __shared__ _Float16 Bs[3][128 * 40];
    const int tid = threadIdx.x, wave = tid >> 5, lane = tid & 31;
    const int wm = wave & 3, wn = wave >> 2;
    const int hi = lane >> 4, ln = lane & 15;
    const int m0 = blockIdx.x * 128, n0 = blockIdx.y * 128;

    v8f acc[2][4];
#pragma unroll
    for (int i = 0; i < 2; ++i)
#pragma unroll
        for (int j = 0; j < 4; ++j) acc[i][j] = vzero();

    GEMM_MAINLOOP(1024, 1024)

#pragma unroll
    for (int mi = 0; mi < 2; ++mi)
#pragma unroll
        for (int ni = 0; ni < 4; ++ni)
#pragma unroll
            for (int e = 0; e < 8; ++e) {
                const int gm = m0 + wm * 32 + mi * 16 + hi * 8 + e;
                const int gn = n0 + wn * 64 + ni * 16 + ln;
                out[(size_t)gm * DOUT + gn] =
                    xin[(size_t)gm * DOUT + gn] * res_scale[gn] +
                    acc[mi][ni][e] * layer_scale[gn];
            }
}

// ---------------------------------------------------------------------------
// Launcher
// ---------------------------------------------------------------------------
extern "C" void kernel_launch(void* const* d_in, const int* in_sizes, int n_in,
                              void* d_out, int out_size, void* d_ws, size_t ws_size,
                              hipStream_t stream) {
    (void)in_sizes; (void)n_in; (void)out_size; (void)ws_size;
    const float* x           = (const float*)d_in[0];
    const float* uv_w        = (const float*)d_in[1];
    const float* o_w         = (const float*)d_in[2];
    const float* gamma       = (const float*)d_in[3];
    const float* beta        = (const float*)d_in[4];
    const float* wbias       = (const float*)d_in[5];
    const float* g           = (const float*)d_in[6];
    const float* res_scale   = (const float*)d_in[7];
    const float* layer_scale = (const float*)d_in[8];
    float* out = (float*)d_out;

    char* ws = (char*)d_ws;
    size_t off = 0;
    auto take = [&](size_t bytes) { char* p = ws + off; off += (bytes + 255) & ~(size_t)255; return p; };
    _Float16* xn_h   = (_Float16*)take((size_t)MTOT * DIN * 2);
    _Float16* uvw_h  = (_Float16*)take((size_t)UVN * DIN * 2);
    _Float16* ow_h   = (_Float16*)take((size_t)DOUT * EDIM * 2);
    _Float16* u_h    = (_Float16*)take((size_t)MTOT * EDIM * 2);
    _Float16* v_h    = (_Float16*)take((size_t)MTOT * EDIM * 2);
    float*    base_f = (float*)take((size_t)MTOT * SDIM * 4);
    _Float16* q_h    = (_Float16*)take((size_t)MTOT * SDIM * 2);
    _Float16* k_h    = (_Float16*)take((size_t)MTOT * SDIM * 2);
    _Float16* attn_h = (_Float16*)take((size_t)MTOT * EDIM * 2);

    pgau_norm_shift<<<MTOT, 256, 0, stream>>>(x, g, xn_h);
    pgau_f32_to_f16<<<(UVN * DIN + 255) / 256, 256, 0, stream>>>(uv_w, uvw_h, UVN * DIN);
    pgau_f32_to_f16<<<(DOUT * EDIM + 255) / 256, 256, 0, stream>>>(o_w, ow_h, DOUT * EDIM);
    pgau_gemm_uv<<<dim3(MTOT / 128, UVN / 128), 256, 0, stream>>>(xn_h, uvw_h, u_h, v_h, base_f);
    pgau_rope_qk<<<MTOT, 64, 0, stream>>>(base_f, gamma, beta, q_h, k_h);
    pgau_attn<<<dim3(LSEQ / 64, EDIM / 256, BATCH), 256, 0, stream>>>(q_h, k_h, v_h, u_h, wbias, attn_h);
    pgau_gemm_out<<<dim3(MTOT / 128, DOUT / 128), 256, 0, stream>>>(attn_h, ow_h, x, res_scale, layer_scale, out);
}
